// TacticalGNN_55439437857003
// MI455X (gfx1250) — matrix operand order repaired and checked by reading, not compiled, not run on hardware
//
#include <hip/hip_runtime.h>
#include <math.h>

#define N_NODES 16384
#define N_EDGES 131072
#define N_GRAPH 64
#define N_PAIRS 8192

typedef __bf16 bf16;
typedef __attribute__((ext_vector_type(16))) __bf16 v16bf;
typedef __attribute__((ext_vector_type(8)))  __bf16 v8bf;   // 16 bytes
typedef __attribute__((ext_vector_type(8)))  float  v8f;

// ---------------- device helpers ----------------
__device__ __forceinline__ float geluf(float x) { return 0.5f * x * (1.f + erff(x * 0.70710678118654752f)); }
__device__ __forceinline__ float eluf(float x)  { return x > 0.f ? x : expm1f(x); }
__device__ __forceinline__ float sigm(float x)  { return 1.f / (1.f + expf(-x)); }
__device__ __forceinline__ unsigned f2ord(float f) {
  unsigned u = __float_as_uint(f);
  return (u & 0x80000000u) ? ~u : (u | 0x80000000u);
}
__device__ __forceinline__ float ord2f(unsigned u) {
  return __uint_as_float((u & 0x80000000u) ? (u & 0x7FFFFFFFu) : ~u);
}

// ---------------- generic fills ----------------
__global__ void fill_f32(float* p, float v, int n) {
  int i = blockIdx.x * blockDim.x + threadIdx.x;
  if (i < n) p[i] = v;
}
__global__ void fill_u32(unsigned* p, unsigned v, int n) {
  int i = blockIdx.x * blockDim.x + threadIdx.x;
  if (i < n) p[i] = v;
}

// ---------------- weight swizzle: f32 [K,Nc] row-major -> bf16 WMMA-B per-lane layout ----------------
// Bs[(((tN*(K/32) + kc)*32 + lane)*16 + idx] = B[kc*32 + (idx>>3)*16 + (lane>>4)*8 + (idx&7), tN*16 + (lane&15)]
__global__ void swizzle_b(const float* __restrict__ B, bf16* __restrict__ Bs, int K, int Nc) {
  int i = blockIdx.x * blockDim.x + threadIdx.x;
  if (i >= K * Nc) return;
  int idx  = i & 15;
  int lane = (i >> 4) & 31;
  int rest = i >> 9;                 // tN*(K/32) + kc
  int KC = K >> 5;
  int tN = rest / KC, kc = rest - tN * KC;
  int kk  = (kc << 5) + ((idx >> 3) << 4) + ((lane >> 4) << 3) + (idx & 7);
  int col = tN * 16 + (lane & 15);
  Bs[i] = (bf16)B[(size_t)kk * Nc + col];
}

// ---------------- BatchNorm ----------------
__global__ void bn_stats(const float* __restrict__ x, float* mu, float* var, int N, int F) {
  int f = blockIdx.x, t = threadIdx.x;
  float s = 0.f, q = 0.f;
  for (int n = t; n < N; n += blockDim.x) { float v = x[n * F + f]; s += v; q += v * v; }
  __shared__ float ss[256], sq[256];
  ss[t] = s; sq[t] = q; __syncthreads();
  for (int off = 128; off > 0; off >>= 1) {
    if (t < off) { ss[t] += ss[t + off]; sq[t] += sq[t + off]; }
    __syncthreads();
  }
  if (t == 0) { float m = ss[0] / N; mu[f] = m; var[f] = sq[0] / N - m * m; }
}
__global__ void bn_apply(const float* __restrict__ x, const float* mu, const float* var,
                         const float* g, const float* b, float* x0, int total, int F) {
  int i = blockIdx.x * blockDim.x + threadIdx.x;
  if (i >= total) return;
  int f = i % F;
  x0[i] = (x[i] - mu[f]) * rsqrtf(var[f] + 1e-5f) * g[f] + b[f];
}

// ---------------- layer-1 projections (K=11, scalar) ----------------
__global__ void proj_l1(const float* __restrict__ x0,
                        const float* __restrict__ wl, const float* __restrict__ bl,
                        const float* __restrict__ wr, const float* __restrict__ br,
                        float* __restrict__ xl, float* __restrict__ xr) {
  __shared__ float xs[11];
  int n = blockIdx.x, t = threadIdx.x; // blockDim 384
  if (t < 11) xs[t] = x0[n * 11 + t];
  __syncthreads();
  float sl = bl[t], sr = br[t];
#pragma unroll
  for (int k = 0; k < 11; ++k) { float v = xs[k]; sl += v * wl[k * 384 + t]; sr += v * wr[k * 384 + t]; }
  xl[n * 384 + t] = sl; xr[n * 384 + t] = sr;
}
__global__ void proj_resid(const float* __restrict__ x0, const float* __restrict__ w,
                           const float* __restrict__ b, float* __restrict__ resid) {
  __shared__ float xs[11];
  int n = blockIdx.x, t = threadIdx.x; // blockDim 96
  if (t < 11) xs[t] = x0[n * 11 + t];
  __syncthreads();
  float s = b[t];
#pragma unroll
  for (int k = 0; k < 11; ++k) s += xs[k] * w[k * 96 + t];
  resid[n * 96 + t] = s;
}

// ---------------- GATv2 edge passes ----------------
__global__ void edge_a(const int* __restrict__ src, const int* __restrict__ dst,
                       const float* __restrict__ ea,
                       const float* __restrict__ xl, const float* __restrict__ xr,
                       const float* __restrict__ we, const float* __restrict__ att,
                       float* __restrict__ aout, unsigned* __restrict__ amax,
                       int E, int H, int C) {
  int gid = blockIdx.x * blockDim.x + threadIdx.x;
  int w = gid >> 5, lane = gid & 31;
  if (w >= E * H) return;
  int e = w / H, h = w - e * H;
  int s = src[e], d = dst[e];
  int HC = H * C;
  float e0 = ea[e * 5 + 0], e1 = ea[e * 5 + 1], e2 = ea[e * 5 + 2], e3 = ea[e * 5 + 3], e4 = ea[e * 5 + 4];
  float acc = 0.f;
  for (int c = lane; c < C; c += 32) {
    int hc = h * C + c;
    float ev = e0 * we[hc] + e1 * we[HC + hc] + e2 * we[2 * HC + hc] + e3 * we[3 * HC + hc] + e4 * we[4 * HC + hc];
    float m = xl[s * HC + hc] + xr[d * HC + hc] + ev;
    m = m > 0.f ? m : 0.2f * m;           // leaky_relu 0.2
    acc += m * att[hc];
  }
  for (int o = 16; o > 0; o >>= 1) acc += __shfl_xor(acc, o, 32);
  if (lane == 0) {
    aout[w] = acc;
    atomicMax(&amax[d * H + h], f2ord(acc));
  }
}
__global__ void edge_b(const int* __restrict__ dst, float* __restrict__ ap,
                       const unsigned* __restrict__ amax, float* __restrict__ den, int E, int H) {
  int i = blockIdx.x * blockDim.x + threadIdx.x;
  if (i >= E * H) return;
  int e = i / H, h = i - e * H;
  int d = dst[e];
  float p = expf(ap[i] - ord2f(amax[d * H + h]));
  ap[i] = p;
  atomicAdd(&den[d * H + h], p);
}
__global__ void edge_c(const int* __restrict__ src, const int* __restrict__ dst,
                       const float* __restrict__ ap, const float* __restrict__ xl,
                       float* __restrict__ out, int E, int H, int C) {
  int gid = blockIdx.x * blockDim.x + threadIdx.x;
  int w = gid >> 5, lane = gid & 31;
  if (w >= E * H) return;
  int e = w / H, h = w - e * H;
  int s = src[e], d = dst[e];
  int HC = H * C;
  float p = ap[w];
  for (int c = lane; c < C; c += 32)
    atomicAdd(&out[d * HC + h * C + c], p * xl[s * HC + h * C + c]);
}

// ---------------- node finalize: /den + bias -> LN -> ELU ----------------
__global__ void fin_l1(const float* __restrict__ out1, const float* __restrict__ den,
                       const float* __restrict__ bias, const float* __restrict__ lg,
                       const float* __restrict__ lb, bf16* __restrict__ h1bf) {
  int n = blockIdx.x, t = threadIdx.x; // blockDim 384
  __shared__ float sm[512];
  float v = out1[n * 384 + t] / (den[n * 4 + t / 96] + 1e-16f) + bias[t];
  sm[t] = v; if (t < 128) sm[384 + t] = 0.f;
  __syncthreads();
  for (int off = 256; off > 0; off >>= 1) { if (t < off) sm[t] += sm[t + off]; __syncthreads(); }
  float mu = sm[0] / 384.f; __syncthreads();
  float dv = v - mu;
  sm[t] = dv * dv; if (t < 128) sm[384 + t] = 0.f;
  __syncthreads();
  for (int off = 256; off > 0; off >>= 1) { if (t < off) sm[t] += sm[t + off]; __syncthreads(); }
  float var = sm[0] / 384.f;
  float y = eluf(dv * rsqrtf(var + 1e-5f) * lg[t] + lb[t]);
  h1bf[n * 384 + t] = (bf16)y;
}
__global__ void fin_l2(const float* __restrict__ out2, const float* __restrict__ den,
                       const float* __restrict__ bias, const float* __restrict__ lg,
                       const float* __restrict__ lb, const float* __restrict__ resid,
                       float* __restrict__ node, bf16* __restrict__ nodebf) {
  int n = blockIdx.x, t = threadIdx.x; // blockDim 96
  __shared__ float sm[128];
  float v = out2[n * 96 + t] / (den[n] + 1e-16f) + bias[t];
  sm[t] = v; if (t < 32) sm[96 + t] = 0.f;
  __syncthreads();
  for (int off = 64; off > 0; off >>= 1) { if (t < off) sm[t] += sm[t + off]; __syncthreads(); }
  float mu = sm[0] / 96.f; __syncthreads();
  float dv = v - mu;
  sm[t] = dv * dv; if (t < 32) sm[96 + t] = 0.f;
  __syncthreads();
  for (int off = 64; off > 0; off >>= 1) { if (t < off) sm[t] += sm[t + off]; __syncthreads(); }
  float var = sm[0] / 96.f;
  float y = eluf(dv * rsqrtf(var + 1e-5f) * lg[t] + lb[t]) + resid[n * 96 + t];
  node[n * 96 + t] = y;
  nodebf[n * 96 + t] = (bf16)y;
}

// ---------------- WMMA bf16 GEMM: C[M,Nc] = A[M,K] * B[K,Nc] + bias ----------------
// A: row-major bf16 (rows 16B-aligned). Bs: pre-swizzled per-lane layout (swizzle_b).
// One 16x16 output tile per wave; K templated so the WMMA chain fully unrolls.
template <int K>
__global__ void gemm_bf16(const bf16* __restrict__ A, const bf16* __restrict__ Bs,
                          const float* __restrict__ bias, float* __restrict__ C,
                          int M, int Nc) {
  int gid = blockIdx.x * blockDim.x + threadIdx.x;
  int wave = gid >> 5, lane = gid & 31;
  int tilesN = Nc >> 4;
  int nTiles = (M >> 4) * tilesN;
  if (wave >= nTiles) return;
  int tM = wave / tilesN, tN = wave - tM * tilesN;
  int row = tM * 16 + (lane & 15);
  int hi = lane >> 4;
  constexpr int KC = K / 32;
  const v8bf* Ap = (const v8bf*)(A + (size_t)row * K + (hi << 3)); // 16B-aligned
  const v8bf* Bp = (const v8bf*)(Bs + ((size_t)(tN * KC) * 32 + lane) * 16);
  v8f acc = {0.f, 0.f, 0.f, 0.f, 0.f, 0.f, 0.f, 0.f};
#pragma unroll
  for (int kc = 0; kc < KC; ++kc) {
    union { v16bf v; v8bf h[2]; } ua, ub;
    // A: lane holds K = kc*32 + hi*8 + [0..7] and kc*32 + 16 + hi*8 + [0..7]
    ua.h[0] = Ap[kc * 4];       // +32 bf16 per kc = 4 x v8bf
    ua.h[1] = Ap[kc * 4 + 2];   // +16 bf16
    // B: contiguous 16 bf16 per lane per chunk
    ub.h[0] = Bp[kc * 64];      // chunk stride = 32 lanes * 16 bf16 = 64 x v8bf
    ub.h[1] = Bp[kc * 64 + 1];
    acc = __builtin_amdgcn_wmma_f32_16x16x32_bf16(false, ua.v, false, ub.v, (short)0, acc, false, false);
  }
#pragma unroll
  for (int v = 0; v < 8; ++v) {
    int r = tM * 16 + hi * 8 + v;
    int c = tN * 16 + (lane & 15);
    C[(size_t)r * Nc + c] = acc[v] + bias[c];
  }
}

// ---------------- pooling ----------------
__global__ void pool_atomic(const float* __restrict__ node, const int* __restrict__ batch,
                            float* gsum, unsigned* gmax, float* cnt) {
  int i = blockIdx.x * blockDim.x + threadIdx.x;
  if (i >= N_NODES * 96) return;
  int n = i / 96, c = i - n * 96;
  int b = batch[n];
  float v = node[i];
  atomicAdd(&gsum[b * 96 + c], v);
  atomicMax(&gmax[b * 96 + c], f2ord(v));
  if (c == 0) atomicAdd(&cnt[b], 1.f);
}
__global__ void pool_final(const float* gsum, const unsigned* gmax, const float* cnt, float* g) {
  int b = blockIdx.x, c = threadIdx.x; // 64 x 96
  g[b * 192 + c]      = gsum[b * 96 + c] / cnt[b];
  g[b * 192 + 96 + c] = ord2f(gmax[b * 96 + c]);
}

// ---------------- GRU step (torch gate order r,z,n) ----------------
__global__ void gru_step(const float* __restrict__ xt, const float* __restrict__ hin,
                         float* __restrict__ hout,
                         const float* __restrict__ wi, const float* __restrict__ wh,
                         const float* __restrict__ bi, const float* __restrict__ bh) {
  __shared__ float xs[192], hs[192];
  int b = blockIdx.x, t = threadIdx.x; // blockDim 192
  xs[t] = xt[b * 192 + t]; hs[t] = hin[b * 192 + t];
  __syncthreads();
  float g0 = bi[t], g1 = bi[192 + t], g2 = bi[384 + t];
  float h0 = bh[t], h1 = bh[192 + t], h2 = bh[384 + t];
  for (int k = 0; k < 192; ++k) {
    float xv = xs[k], hv = hs[k];
    const float* wir = wi + k * 576;
    const float* whr = wh + k * 576;
    g0 += xv * wir[t]; g1 += xv * wir[192 + t]; g2 += xv * wir[384 + t];
    h0 += hv * whr[t]; h1 += hv * whr[192 + t]; h2 += hv * whr[384 + t];
  }
  float r = sigm(g0 + h0);
  float z = sigm(g1 + h1);
  float nn = tanhf(g2 + r * h2);
  hout[b * 192 + t] = (1.f - z) * nn + z * hs[t];
}

// ---------------- t = LN(h + g) ----------------
__global__ void add_ln192(const float* __restrict__ h, const float* __restrict__ g,
                          const float* gam, const float* bet, float* __restrict__ tout) {
  int b = blockIdx.x, t = threadIdx.x; // blockDim 192
  __shared__ float sm[256];
  float v = h[b * 192 + t] + g[b * 192 + t];
  sm[t] = v; if (t < 64) sm[192 + t] = 0.f;
  __syncthreads();
  for (int off = 128; off > 0; off >>= 1) { if (t < off) sm[t] += sm[t + off]; __syncthreads(); }
  float mu = sm[0] / 192.f; __syncthreads();
  float dv = v - mu;
  sm[t] = dv * dv; if (t < 64) sm[192 + t] = 0.f;
  __syncthreads();
  for (int off = 128; off > 0; off >>= 1) { if (t < off) sm[t] += sm[t + off]; __syncthreads(); }
  float var = sm[0] / 192.f;
  tout[b * 192 + t] = dv * rsqrtf(var + 1e-5f) * gam[t] + bet[t];
}

// ---------------- change-head input concat ----------------
__global__ void build_cgin(const float* g, const float* pg, float* cgin) {
  int i = blockIdx.x * blockDim.x + threadIdx.x;
  if (i >= 64 * 576) return;
  int b = i / 576, c = i - b * 576;
  float v;
  if (c < 192) v = g[b * 192 + c];
  else if (c < 384) v = pg[b * 192 + c - 192];
  else v = g[b * 192 + c - 384] - pg[b * 192 + c - 384];
  cgin[i] = v;
}

// ---------------- generic head: LN(GELU(in@aw+ab)) -> GELU(@w1+b1) -> @w2+b2 ----------------
__global__ void head_mlp(const float* __restrict__ in, int Din,
                         const float* aw, const float* ab, const float* ag, const float* abt, int Da,
                         const float* w1, const float* b1, int Dh,
                         const float* w2, const float* b2, int Do,
                         float* __restrict__ out) {
  __shared__ float sin_[576], sa[192], sh1[96], sred[256];
  int b = blockIdx.x, t = threadIdx.x; // blockDim 192
  for (int i = t; i < Din; i += 192) sin_[i] = in[b * Din + i];
  __syncthreads();
  float av = 0.f;
  if (t < Da) {
    av = ab[t];
    for (int k = 0; k < Din; ++k) av += sin_[k] * aw[k * Da + t];
    av = geluf(av);
  }
  sred[t] = (t < Da) ? av : 0.f; if (t < 64) sred[192 + t] = 0.f;
  __syncthreads();
  for (int off = 128; off > 0; off >>= 1) { if (t < off) sred[t] += sred[t + off]; __syncthreads(); }
  float mu = sred[0] / Da; __syncthreads();
  float dv = av - mu;
  sred[t] = (t < Da) ? dv * dv : 0.f; if (t < 64) sred[192 + t] = 0.f;
  __syncthreads();
  for (int off = 128; off > 0; off >>= 1) { if (t < off) sred[t] += sred[t + off]; __syncthreads(); }
  float var = sred[0] / Da;
  if (t < Da) sa[t] = dv * rsqrtf(var + 1e-5f) * ag[t] + abt[t];
  __syncthreads();
  if (t < Dh) {
    float v = b1[t];
    for (int k = 0; k < Da; ++k) v += sa[k] * w1[k * Dh + t];
    sh1[t] = geluf(v);
  }
  __syncthreads();
  if (t < Do) {
    float v = b2[t];
    for (int k = 0; k < Dh; ++k) v += sh1[k] * w2[k * Do + t];
    out[b * Do + t] = v;
  }
}

// ---------------- link head ----------------
__global__ void lngelu96(float* __restrict__ a, const float* g, const float* b) {
  int n = blockIdx.x, t = threadIdx.x; // blockDim 96
  __shared__ float sm[128];
  float v = geluf(a[n * 96 + t]);
  sm[t] = v; if (t < 32) sm[96 + t] = 0.f;
  __syncthreads();
  for (int off = 64; off > 0; off >>= 1) { if (t < off) sm[t] += sm[t + off]; __syncthreads(); }
  float mu = sm[0] / 96.f; __syncthreads();
  float dv = v - mu;
  sm[t] = dv * dv; if (t < 32) sm[96 + t] = 0.f;
  __syncthreads();
  for (int off = 64; off > 0; off >>= 1) { if (t < off) sm[t] += sm[t + off]; __syncthreads(); }
  float var = sm[0] / 96.f;
  a[n * 96 + t] = dv * rsqrtf(var + 1e-5f) * g[t] + b[t];
}
__global__ void linkpair(const float* __restrict__ ad, const int* __restrict__ lp,
                         const float* __restrict__ x,
                         const float* w1, const float* b1, const float* w2, const float* b2,
                         const float* w3, const float* b3, const float* comp,
                         float* __restrict__ out) {
  __shared__ float pe[192], sh1[96], sh2[48], sred[128];
  int p = blockIdx.x, t = threadIdx.x; // blockDim 96
  int i = lp[p * 2 + 0], j = lp[p * 2 + 1];
  pe[t] = ad[i * 96 + t];
  pe[96 + t] = ad[j * 96 + t];
  __syncthreads();
  float v = b1[t];
  for (int k = 0; k < 192; ++k) v += pe[k] * w1[k * 96 + t];
  sh1[t] = geluf(v);
  __syncthreads();
  if (t < 48) {
    float v2 = b2[t];
    for (int k = 0; k < 96; ++k) v2 += sh1[k] * w2[k * 48 + t];
    sh2[t] = geluf(v2);
  }
  __syncthreads();
  sred[t] = (t < 48) ? sh2[t] * w3[t] : 0.f;
  if (t < 32) sred[96 + t] = 0.f;
  __syncthreads();
  for (int off = 64; off > 0; off >>= 1) { if (t < off) sred[t] += sred[t + off]; __syncthreads(); }
  if (t == 0) {
    float emb = sigm(sred[0] + b3[0]);
    const float* xi = x + (size_t)i * 11;
    const float* xj = x + (size_t)j * 11;
    int piv = 0; float mv = xi[6];
    for (int q = 1; q < 5; ++q) if (xi[6 + q] > mv) { mv = xi[6 + q]; piv = q; }
    int pjv = 0; mv = xj[6];
    for (int q = 1; q < 5; ++q) if (xj[6 + q] > mv) { mv = xj[6 + q]; pjv = q; }
    float dx = xi[0] - xj[0], dy = xi[1] - xj[1];
    float d = sqrtf(dx * dx + dy * dy);
    float cmp = sigm(comp[piv * 5 + pjv]) * expf(-d);
    out[p] = 0.7f * emb + 0.3f * cmp;
  }
}

// ---------------- host ----------------
extern "C" void kernel_launch(void* const* d_in, const int* in_sizes, int n_in,
                              void* d_out, int out_size, void* d_ws, size_t ws_size,
                              hipStream_t stream) {
  (void)in_sizes; (void)n_in; (void)out_size; (void)ws_size;
  const float* x   = (const float*)d_in[0];
  const int*   ei  = (const int*)d_in[1];
  const float* ea  = (const float*)d_in[2];
  const int*   bat = (const int*)d_in[3];
  const float* px  = (const float*)d_in[4];
  const int*   pei = (const int*)d_in[5];
  const float* pea = (const float*)d_in[6];
  const int*   pba = (const int*)d_in[7];
  const float* p2x = (const float*)d_in[8];
  const int*   p2ei= (const int*)d_in[9];
  const float* p2ea= (const float*)d_in[10];
  const int*   p2ba= (const int*)d_in[11];
  const int*   lp  = (const int*)d_in[12];
  int k = 13;
#define PF(name) const float* name = (const float*)d_in[k++]
  PF(bn_g); PF(bn_b);
  PF(c1_wl); PF(c1_bl); PF(c1_wr); PF(c1_br); PF(c1_we); PF(c1_att); PF(c1_bias);
  PF(ln1_g); PF(ln1_b);
  PF(c2_wl); PF(c2_bl); PF(c2_wr); PF(c2_br); PF(c2_we); PF(c2_att); PF(c2_bias);
  PF(ln2_g); PF(ln2_b);
  PF(rp_w); PF(rp_b);
  PF(gru_wi); PF(gru_wh); PF(gru_bi); PF(gru_bh);
  PF(tn_g); PF(tn_b);
  PF(th_aw); PF(th_ab); PF(th_ag); PF(th_abt); PF(th_w1); PF(th_b1); PF(th_w2); PF(th_b2);
  PF(ch_aw); PF(ch_ab); PF(ch_ag); PF(ch_abt); PF(ch_w1); PF(ch_b1); PF(ch_w2); PF(ch_b2);
  PF(cg_aw); PF(cg_ab); PF(cg_ag); PF(cg_abt); PF(cg_w1); PF(cg_b1); PF(cg_w2); PF(cg_b2);
  PF(lh_aw); PF(lh_ab); PF(lh_ag); PF(lh_abt); PF(lh_w1); PF(lh_b1); PF(lh_w2); PF(lh_b2); PF(lh_w3); PF(lh_b3);
  PF(comp);
#undef PF

  // ---- workspace bump allocator (deterministic static layout) ----
  char* W = (char*)d_ws;
  size_t off = 0;
  auto AL = [&](size_t bytes) { size_t r = off; off = (off + bytes + 255) & ~(size_t)255; return r; };
  size_t o_wlbf  = AL(384 * 96 * 2);
  size_t o_wrbf  = AL(384 * 96 * 2);
  size_t o_lawbf = AL(96 * 96 * 2);
  size_t o_mu    = AL(11 * 4);
  size_t o_var   = AL(11 * 4);
  size_t o_x0    = AL((size_t)N_NODES * 11 * 4);
  size_t o_regA  = AL((size_t)N_NODES * 384 * 4);  // xl1 ; later xl2/xr2/out2
  size_t o_regB  = AL((size_t)N_NODES * 384 * 4);  // xr1 -> out1 ; later node/ad
  size_t o_h1bf  = AL((size_t)N_NODES * 384 * 2);  // bf16 ; later nodebf
  size_t o_ae    = AL((size_t)N_EDGES * 4 * 4);
  size_t o_amax  = AL((size_t)N_NODES * 4 * 4);
  size_t o_den   = AL((size_t)N_NODES * 4 * 4);
  size_t o_resid = AL((size_t)N_NODES * 96 * 4);
  size_t o_gsum  = AL(64 * 96 * 4);
  size_t o_gmax  = AL(64 * 96 * 4);
  size_t o_cnt   = AL(64 * 4);
  size_t o_g     = AL(64 * 192 * 4);
  size_t o_pg    = AL(64 * 192 * 4);
  size_t o_p2g   = AL(64 * 192 * 4);
  size_t o_hA    = AL(64 * 192 * 4);
  size_t o_hB    = AL(64 * 192 * 4);
  size_t o_t     = AL(64 * 192 * 4);
  size_t o_cgin  = AL(64 * 576 * 4);

  bf16*     wlbf  = (bf16*)(W + o_wlbf);
  bf16*     wrbf  = (bf16*)(W + o_wrbf);
  bf16*     lawbf = (bf16*)(W + o_lawbf);
  float*    mu    = (float*)(W + o_mu);
  float*    var   = (float*)(W + o_var);
  float*    x0    = (float*)(W + o_x0);
  float*    xl1   = (float*)(W + o_regA);
  float*    xr1   = (float*)(W + o_regB);
  float*    out1  = (float*)(W + o_regB);
  float*    xl2   = (float*)(W + o_regA);
  float*    xr2   = (float*)(W + o_regA + (size_t)N_NODES * 96 * 4);
  float*    out2  = (float*)(W + o_regA + (size_t)N_NODES * 96 * 8);
  bf16*     h1bf  = (bf16*)(W + o_h1bf);
  bf16*     nodebf= (bf16*)(W + o_h1bf);
  float*    node  = (float*)(W + o_regB);
  float*    adbuf = (float*)(W + o_regB + (size_t)N_NODES * 96 * 4);
  float*    aebuf = (float*)(W + o_ae);
  unsigned* amax  = (unsigned*)(W + o_amax);
  float*    den   = (float*)(W + o_den);
  float*    resid = (float*)(W + o_resid);
  float*    gsum  = (float*)(W + o_gsum);
  unsigned* gmaxo = (unsigned*)(W + o_gmax);
  float*    cnt   = (float*)(W + o_cnt);
  float*    gbuf  = (float*)(W + o_g);
  float*    pgbuf = (float*)(W + o_pg);
  float*    p2gbuf= (float*)(W + o_p2g);
  float*    hA    = (float*)(W + o_hA);
  float*    hB    = (float*)(W + o_hB);
  float*    tbuf  = (float*)(W + o_t);
  float*    cgin  = (float*)(W + o_cgin);

  float* outp = (float*)d_out;
  float* out_tactic = outp;
  float* out_coarse = outp + 64 * 13;
  float* out_change = outp + 64 * 13 + 64 * 4;
  float* out_link   = outp + 64 * 13 + 64 * 4 + 64;

  auto B256 = [](long n) { return (int)((n + 255) / 256); };

  // prep: weights to bf16 WMMA-B swizzled layout (once)
  swizzle_b<<<B256(384 * 96), 256, 0, stream>>>(c2_wl, wlbf, 384, 96);
  swizzle_b<<<B256(384 * 96), 256, 0, stream>>>(c2_wr, wrbf, 384, 96);
  swizzle_b<<<B256(96 * 96), 256, 0, stream>>>(lh_aw, lawbf, 96, 96);

  auto backbone = [&](const float* xin, const int* eiin, const float* eain, const int* bin, float* gout) {
    const int* src = eiin;
    const int* dst = eiin + N_EDGES;
    // BatchNorm
    bn_stats<<<11, 256, 0, stream>>>(xin, mu, var, N_NODES, 11);
    bn_apply<<<B256((long)N_NODES * 11), 256, 0, stream>>>(xin, mu, var, bn_g, bn_b, x0, N_NODES * 11, 11);
    // L1 projections + residual projection
    proj_l1<<<N_NODES, 384, 0, stream>>>(x0, c1_wl, c1_bl, c1_wr, c1_br, xl1, xr1);
    proj_resid<<<N_NODES, 96, 0, stream>>>(x0, rp_w, rp_b, resid);
    // L1 attention (H=4, C=96)
    fill_u32<<<B256(N_NODES * 4), 256, 0, stream>>>(amax, 0u, N_NODES * 4);
    fill_f32<<<B256(N_NODES * 4), 256, 0, stream>>>(den, 0.f, N_NODES * 4);
    edge_a<<<B256((long)N_EDGES * 4 * 32), 256, 0, stream>>>(src, dst, eain, xl1, xr1, c1_we, c1_att, aebuf, amax, N_EDGES, 4, 96);
    edge_b<<<B256((long)N_EDGES * 4), 256, 0, stream>>>(dst, aebuf, amax, den, N_EDGES, 4);
    fill_f32<<<B256((long)N_NODES * 384), 256, 0, stream>>>(out1, 0.f, N_NODES * 384);
    edge_c<<<B256((long)N_EDGES * 4 * 32), 256, 0, stream>>>(src, dst, aebuf, xl1, out1, N_EDGES, 4, 96);
    fin_l1<<<N_NODES, 384, 0, stream>>>(out1, den, c1_bias, ln1_g, ln1_b, h1bf);
    // L2 projections (WMMA bf16, K=384)
    {
      long waves = (long)(N_NODES / 16) * (96 / 16);
      gemm_bf16<384><<<B256(waves * 32), 256, 0, stream>>>(h1bf, wlbf, c2_bl, xl2, N_NODES, 96);
      gemm_bf16<384><<<B256(waves * 32), 256, 0, stream>>>(h1bf, wrbf, c2_br, xr2, N_NODES, 96);
    }
    // L2 attention (H=1, C=96)
    fill_u32<<<B256(N_NODES), 256, 0, stream>>>(amax, 0u, N_NODES);
    fill_f32<<<B256(N_NODES), 256, 0, stream>>>(den, 0.f, N_NODES);
    edge_a<<<B256((long)N_EDGES * 32), 256, 0, stream>>>(src, dst, eain, xl2, xr2, c2_we, c2_att, aebuf, amax, N_EDGES, 1, 96);
    edge_b<<<B256(N_EDGES), 256, 0, stream>>>(dst, aebuf, amax, den, N_EDGES, 1);
    fill_f32<<<B256((long)N_NODES * 96), 256, 0, stream>>>(out2, 0.f, N_NODES * 96);
    edge_c<<<B256((long)N_EDGES * 32), 256, 0, stream>>>(src, dst, aebuf, xl2, out2, N_EDGES, 1, 96);
    fin_l2<<<N_NODES, 96, 0, stream>>>(out2, den, c2_bias, ln2_g, ln2_b, resid, node, nodebf);
    // pooling
    fill_f32<<<B256(64 * 96), 256, 0, stream>>>(gsum, 0.f, 64 * 96);
    fill_u32<<<B256(64 * 96), 256, 0, stream>>>(gmaxo, 0u, 64 * 96);
    fill_f32<<<1, 64, 0, stream>>>(cnt, 0.f, 64);
    pool_atomic<<<B256((long)N_NODES * 96), 256, 0, stream>>>(node, bin, gsum, gmaxo, cnt);
    pool_final<<<64, 96, 0, stream>>>(gsum, gmaxo, cnt, gout);
  };

  // prev2 -> prev -> current (current's node/nodebf survive for link head)
  backbone(p2x, p2ei, p2ea, p2ba, p2gbuf);
  backbone(px, pei, pea, pba, pgbuf);
  backbone(x, ei, ea, bat, gbuf);

  // GRU over [p2g, pg, g]
  fill_f32<<<B256(64 * 192), 256, 0, stream>>>(hA, 0.f, 64 * 192);
  gru_step<<<64, 192, 0, stream>>>(p2gbuf, hA, hB, gru_wi, gru_wh, gru_bi, gru_bh);
  gru_step<<<64, 192, 0, stream>>>(pgbuf, hB, hA, gru_wi, gru_wh, gru_bi, gru_bh);
  gru_step<<<64, 192, 0, stream>>>(gbuf, hA, hB, gru_wi, gru_wh, gru_bi, gru_bh);
  add_ln192<<<64, 192, 0, stream>>>(hB, gbuf, tn_g, tn_b, tbuf);

  // heads
  head_mlp<<<64, 192, 0, stream>>>(tbuf, 192, th_aw, th_ab, th_ag, th_abt, 96, th_w1, th_b1, 96, th_w2, th_b2, 13, out_tactic);
  head_mlp<<<64, 192, 0, stream>>>(tbuf, 192, ch_aw, ch_ab, ch_ag, ch_abt, 96, ch_w1, ch_b1, 96, ch_w2, ch_b2, 4, out_coarse);
  build_cgin<<<B256(64 * 576), 256, 0, stream>>>(gbuf, pgbuf, cgin);
  head_mlp<<<64, 192, 0, stream>>>(cgin, 576, cg_aw, cg_ab, cg_ag, cg_abt, 192, cg_w1, cg_b1, 96, cg_w2, cg_b2, 1, out_change);

  // link head (WMMA bf16, K=96)
  {
    long waves = (long)(N_NODES / 16) * (96 / 16);
    gemm_bf16<96><<<B256(waves * 32), 256, 0, stream>>>(nodebf, lawbf, lh_ab, adbuf, N_NODES, 96);
  }
  lngelu96<<<N_NODES, 96, 0, stream>>>(adbuf, lh_ag, lh_abt);
  linkpair<<<N_PAIRS, 96, 0, stream>>>(adbuf, lp, x, lh_w1, lh_b1, lh_w2, lh_b2, lh_w3, lh_b3, comp, out_link);
}
